// FeatureAttn3d_67242007986341
// MI455X (gfx1250) — matrix operand order repaired and checked by reading, compiled, not run on hardware
//
#include <hip/hip_runtime.h>

// ---------------- problem constants ----------------
#define Bn    8
#define Cn    256
#define NS    13824          // 24*24*24
#define EMBED 128
#define HEADS 8
#define WORD  64
#define COUT  256
#define OUT0  (Bn * COUT * NS)   // 28,311,552 : offset of attn_out in d_out

// workspace layout (float offsets)
#define XM_OFF   0               // 2048*27
#define QK_OFF   65536           // 2048*256
#define Q_OFF    1048576         // 8*8*256*64
#define K_OFF    2097152         // 8*8*256*64
#define M_OFF    3145728         // 8*256*256            (ends 3,670,016)
#define MSWH_OFF 3670016         // 524288 bf16 (262144 floats)
#define MSWL_OFF 3932160
#define CSWH_OFF 4194304         // 65536 bf16 (32768 floats)
#define CSWL_OFF 4227072         // total ws: 4,259,840 floats = 16.25 MB

typedef __attribute__((ext_vector_type(2)))  float   v2f;
typedef __attribute__((ext_vector_type(8)))  float   v8f;
typedef __attribute__((ext_vector_type(8)))  __bf16  v8bf;
typedef __attribute__((ext_vector_type(16))) __bf16  v16bf;
typedef __attribute__((ext_vector_type(4)))  unsigned int u32x4;
typedef __attribute__((ext_vector_type(8)))  int     i32x8;
typedef __attribute__((ext_vector_type(4)))  int     i32x4;

// V_WMMA_F32_16X16X4_F32 (exact f32, used for attention scores)
#define WMMA4(a, b, c) \
    __builtin_amdgcn_wmma_f32_16x16x4_f32(false, (a), false, (b), (short)0, (c), false, false)
// V_WMMA_F32_16X16X32_BF16 (full-rate path for the big GEMMs)
#define WMMAB(a, b, c) \
    __builtin_amdgcn_wmma_f32_16x16x32_bf16(false, (a), false, (b), (short)0, (c), false, false)

#if defined(__HIP_DEVICE_COMPILE__) && __has_builtin(__builtin_amdgcn_tensor_load_to_lds)
#define TDM_OK 1
#else
#define TDM_OK 0
#endif

__device__ __forceinline__ void bsplit(float x, __bf16& hi, __bf16& lo) {
    hi = (__bf16)x;
    lo = (__bf16)(x - (float)hi);
}

// ---------------- K1: 8x8x8 block-mean pooling ----------------
__global__ __launch_bounds__(256) void k_pool(const float* __restrict__ x,
                                              float* __restrict__ xmean) {
    __shared__ float cell[27];
    const int bc  = blockIdx.x;            // b*256 + c
    const int tid = threadIdx.x;
    if (tid < 27) cell[tid] = 0.0f;
    __syncthreads();
    const float* xp = x + (size_t)bc * NS;
    float part[27];
#pragma unroll
    for (int i = 0; i < 27; ++i) part[i] = 0.0f;
    for (int i = tid; i < NS; i += 256) {
        int d = i / 576, rem = i % 576;
        int w = rem / 24, h = rem % 24;
        int cidx = (d >> 3) * 9 + (w >> 3) * 3 + (h >> 3);
        part[cidx] += xp[i];
    }
#pragma unroll
    for (int i = 0; i < 27; ++i) atomicAdd(&cell[i], part[i]);   // ds_add_f32
    __syncthreads();
    if (tid < 27) xmean[bc * 27 + tid] = cell[tid] * (1.0f / 512.0f);
}

// ---------------- K2a: qk = xmean @ qk_w^T + qk_b  (K=27) ----------------
__global__ __launch_bounds__(256) void k_qk(const float* __restrict__ xmean,
                                            const float* __restrict__ qk_w,
                                            const float* __restrict__ qk_b,
                                            float* __restrict__ qk) {
    __shared__ float row[27];
    const int bc = blockIdx.x, j = threadIdx.x;
    if (j < 27) row[j] = xmean[bc * 27 + j];
    __syncthreads();
    float acc = qk_b[j];
    const float* wp = qk_w + j * 27;
#pragma unroll
    for (int p = 0; p < 27; ++p) acc += row[p] * wp[p];
    qk[bc * 256 + j] = acc;
}

// ---------------- K2b: q/k head projections (K=128) ----------------
__global__ __launch_bounds__(256) void k_heads(const float* __restrict__ qk,
                                               const float* __restrict__ wq_w,
                                               const float* __restrict__ wq_b,
                                               const float* __restrict__ wk_w,
                                               const float* __restrict__ wk_b,
                                               float* __restrict__ qout,
                                               float* __restrict__ kout) {
    __shared__ float row[256];
    const int bc = blockIdx.x, tid = threadIdx.x;
    const int b = bc >> 8, c = bc & 255;
    row[tid] = qk[bc * 256 + tid];
    __syncthreads();
#pragma unroll
    for (int rep = 0; rep < 2; ++rep) {
        int j = rep * 256 + tid;            // 0..511
        int h = j >> 6, w = j & 63;
        float aq = wq_b[j], ak = wk_b[j];
        const float* wq = wq_w + j * EMBED;
        const float* wk = wk_w + j * EMBED;
        for (int p = 0; p < EMBED; ++p) {
            aq += row[p] * wq[p];
            ak += row[128 + p] * wk[p];
        }
        int dst = (((b * HEADS + h) * Cn) + c) * WORD + w;
        qout[dst] = aq;
        kout[dst] = ak;
    }
}

// ---------------- K0: zero M accumulator ----------------
__global__ __launch_bounds__(256) void k_zero(float* __restrict__ m) {
    m[blockIdx.x * 256 + threadIdx.x] = 0.0f;
}

// ---------------- K3: scores (f32 WMMA) + softmax + M accumulation ----------
// block = (b, h, row-block of 64); dyn LDS = 64*64 + 256*64 + 64*256 floats = 144KB
__global__ __launch_bounds__(256) void k_attn(const float* __restrict__ q,
                                              const float* __restrict__ k,
                                              const float* __restrict__ fch_w,
                                              float* __restrict__ M) {
    extern __shared__ float smem[];
    float* qs  = smem;                // [64][64]
    float* ks  = smem + 4096;         // [256][64]
    float* ssh = smem + 4096 + 16384; // [64][256]

    const int blk = blockIdx.x;
    const int b = blk >> 5, h = (blk >> 2) & 7, rb = blk & 3;
    const int tid  = threadIdx.x;
    const int lane = tid & 31, wave = tid >> 5;
    const int lm = lane & 15, half = lane >> 4;

    const size_t qkbase = (size_t)(b * HEADS + h) * Cn * WORD;
    for (int i = tid; i < 4096; i += 256)
        qs[i] = q[qkbase + (size_t)(rb * 64) * WORD + i];
    for (int i = tid; i < 16384; i += 256)
        ks[i] = k[qkbase + i];
    __syncthreads();

    const int rt = wave & 3;
    const int ctb = (wave >> 2) * 8;
    v8f acc[8];
#pragma unroll
    for (int i = 0; i < 8; ++i) acc[i] = v8f{};
    const float* arow = qs + (rt * 16 + lm) * 64 + 2 * half;
    for (int kk = 0; kk < WORD; kk += 4) {
        v2f a = *(const v2f*)(arow + kk);
#pragma unroll
        for (int c8 = 0; c8 < 8; ++c8) {
            const float* brow = ks + ((ctb + c8) * 16 + lm) * 64 + 2 * half + kk;
            v2f bf = *(const v2f*)brow;
            acc[c8] = WMMA4(a, bf, acc[c8]);
        }
    }
#pragma unroll
    for (int c8 = 0; c8 < 8; ++c8)
#pragma unroll
        for (int i = 0; i < 8; ++i)
            ssh[(rt * 16 + i + 8 * half) * 256 + (ctb + c8) * 16 + lm] = acc[c8][i] * 0.125f;
    __syncthreads();

    const float fw = fch_w[h];
    float* Mrow0 = M + ((size_t)b * Cn + rb * 64) * Cn;
    for (int rr = 0; rr < 8; ++rr) {
        int r = wave * 8 + rr;
        float* srow = ssh + r * 256;
        float mx = -3.4e38f;
        for (int j = lane; j < 256; j += 32) mx = fmaxf(mx, srow[j]);
#pragma unroll
        for (int off = 16; off > 0; off >>= 1) mx = fmaxf(mx, __shfl_xor(mx, off, 32));
        float sum = 0.0f;
        for (int j = lane; j < 256; j += 32) {
            float e = __expf(srow[j] - mx);
            srow[j] = e;
            sum += e;
        }
#pragma unroll
        for (int off = 16; off > 0; off >>= 1) sum += __shfl_xor(sum, off, 32);
        float sc = fw / sum;
        for (int j = lane; j < 256; j += 32)
            atomicAdd(&Mrow0[(size_t)r * Cn + j], srow[j] * sc);   // global_atomic_add_f32
    }
}

// ---------------- K4: attn_out = M + fch_b ----------------
__global__ __launch_bounds__(256) void k_attnout(const float* __restrict__ M,
                                                 const float* __restrict__ fch_b,
                                                 float* __restrict__ out) {
    int idx = blockIdx.x * 256 + threadIdx.x;
    out[OUT0 + idx] = M[idx] + fch_b[0];
}

// ---------------- K6: split 256x256 f32 matrices into swizzled bf16 hi/lo ----
// A[row][k] (per 65536-elem matrix) -> hi/lo at mat*65536 + ((k>>3)*256+row)*8 + (k&7)
// so an A-fragment (8 consecutive k of one row) is one contiguous 16B load.
__global__ __launch_bounds__(256) void k_splitA(const float* __restrict__ src,
                                                __bf16* __restrict__ hi,
                                                __bf16* __restrict__ lo) {
    int idx = blockIdx.x * 256 + threadIdx.x;
    int mat = idx >> 16, rem = idx & 65535;
    int row = rem >> 8, kk = rem & 255;
    float v = src[idx];
    __bf16 h, l;
    bsplit(v, h, l);
    size_t dst = (size_t)mat * 65536 + (((kk >> 3) * 256 + row) * 8) + (kk & 7);
    hi[dst] = h;
    lo[dst] = l;
}

// ---------------- K5: fused  z = M@v + x + fch_b ;  out = conv_w@z + conv_b ----
// bf16 hi/lo split GEMMs (3x v_wmma_f32_16x16x32_bf16 per 16x16x32 step).
// block = (b, 64-wide n tile); dyn LDS = 192KB:
//   vt   f32  [256][64]                     (filled by TDM tensor_load_to_lds)
//   vtsh/vtsl bf16 swizzled: (k,n) -> ((k>>4)*64+n)*16 + (k&15)   (32B B-frags)
//   ztsh/ztsl bf16 swizzled, same layout
__global__ __launch_bounds__(256) void k_biggemm(const float* __restrict__ x,
                                                 const __bf16* __restrict__ Mh,
                                                 const __bf16* __restrict__ Ml,
                                                 const __bf16* __restrict__ Ch,
                                                 const __bf16* __restrict__ Cl,
                                                 const float* __restrict__ conv_b,
                                                 const float* __restrict__ fch_b,
                                                 float* __restrict__ out) {
    extern __shared__ float smem[];
    float*  vt   = smem;                      // 64KB
    __bf16* vtsh = (__bf16*)(smem + 16384);   // 32KB
    __bf16* vtsl = (__bf16*)(smem + 24576);   // 32KB
    __bf16* ztsh = (__bf16*)(smem + 32768);   // 32KB
    __bf16* ztsl = (__bf16*)(smem + 40960);   // 32KB

    const int blk = blockIdx.x;
    const int b = blk / 216;
    const int n0 = (blk % 216) * 64;
    const int tid  = threadIdx.x;
    const int lane = tid & 31, wave = tid >> 5;
    const int lm = lane & 15, half = lane >> 4;
    const float fchb = fch_b[0];

    const float* xtile = x + (size_t)b * Cn * NS + n0;

#if TDM_OK
    // ---- Tensor Data Mover: 2D tile (64 x 256 f32, row stride NS) -> LDS ----
    if (wave == 0) {
        unsigned long long ga = (unsigned long long)(uintptr_t)xtile;
        unsigned int       la = (unsigned int)(uintptr_t)vt;      // LDS byte offset
        u32x4 g0 = { 1u,                                   // count=1, user mode
                     la,                                   // lds_addr
                     (unsigned int)ga,                     // global_addr[31:0]
                     ((unsigned int)(ga >> 32) & 0x01FFFFFFu) | (2u << 30) }; // [56:32] | type=2
        i32x8 g1 = { (int)(2u << 16),                      // data_size = 4B
                     (int)(13824u << 16),                  // tensor_dim0[15:0]
                     (int)(256u << 16),                    // tensor_dim1[15:0]
                     (int)(64u << 16),                     // tile_dim0 = 64
                     256,                                  // tile_dim1 = 256
                     13824,                                // tensor_dim0_stride[31:0]
                     0, 0 };
        i32x4 gz4 = { 0, 0, 0, 0 };
        i32x8 gz8 = { 0, 0, 0, 0, 0, 0, 0, 0 };
        // clang-23 / amdgpu-toolchain 6-arg form:
        // (u32x4 g0, i32x8 g1, i32x4 g2, i32x4 g3, i32x8, i32 cpol)
        __builtin_amdgcn_tensor_load_to_lds(g0, g1, gz4, gz4, gz8, 0);
        __builtin_amdgcn_s_wait_tensorcnt(0);
    }
#else
    for (int idx = tid; idx < 16384; idx += 256) {
        int r = idx >> 6, cidx = idx & 63;
        vt[idx] = xtile[(size_t)r * NS + cidx];
    }
#endif
    __syncthreads();

    // ---- split v tile into swizzled bf16 hi/lo ----
    for (int idx = tid; idx < 16384; idx += 256) {
        int kk = idx >> 6, n = idx & 63;
        __bf16 h, l;
        bsplit(vt[idx], h, l);
        int dst = (((kk >> 4) * 64 + n) * 16) + (kk & 15);
        vtsh[dst] = h;
        vtsl[dst] = l;
    }
    __syncthreads();

    // ---- GEMM1: z = M[b] @ v  (+ x + fch_b), bf16-split ----
    const __bf16* Mhb = Mh + (size_t)b * 65536;
    const __bf16* Mlb = Ml + (size_t)b * 65536;
#pragma unroll
    for (int rh = 0; rh < 2; ++rh) {
        const int R = (wave * 2 + rh) * 16;
        const int row = R + lm;
        v8f acc[4];
#pragma unroll
        for (int i = 0; i < 4; ++i) acc[i] = v8f{};
        for (int k0 = 0; k0 < Cn; k0 += 32) {
            const int o1 = (k0 >> 3) + 2 * half, o2 = o1 + 1;
            v8bf h1 = *(const v8bf*)(Mhb + ((size_t)o1 * 256 + row) * 8);
            v8bf h2 = *(const v8bf*)(Mhb + ((size_t)o2 * 256 + row) * 8);
            v8bf l1 = *(const v8bf*)(Mlb + ((size_t)o1 * 256 + row) * 8);
            v8bf l2 = *(const v8bf*)(Mlb + ((size_t)o2 * 256 + row) * 8);
            v16bf ah = __builtin_shufflevector(h1, h2, 0,1,2,3,4,5,6,7,8,9,10,11,12,13,14,15);
            v16bf al = __builtin_shufflevector(l1, l2, 0,1,2,3,4,5,6,7,8,9,10,11,12,13,14,15);
            const int kb = ((k0 >> 4) + half) * 64;
#pragma unroll
            for (int ct = 0; ct < 4; ++ct) {
                const int n = ct * 16 + lm;
                v16bf bh = *(const v16bf*)(vtsh + (size_t)(kb + n) * 16);
                v16bf bl = *(const v16bf*)(vtsl + (size_t)(kb + n) * 16);
                acc[ct] = WMMAB(ah, bh, acc[ct]);
                acc[ct] = WMMAB(ah, bl, acc[ct]);
                acc[ct] = WMMAB(al, bh, acc[ct]);
            }
        }
#pragma unroll
        for (int ct = 0; ct < 4; ++ct)
#pragma unroll
            for (int i = 0; i < 8; ++i) {
                int m = R + i + 8 * half, n = ct * 16 + lm;
                float z = acc[ct][i] + vt[m * 64 + n] + fchb;
                __bf16 zh, zl;
                bsplit(z, zh, zl);
                int dst = (((m >> 4) * 64 + n) * 16) + (m & 15);
                ztsh[dst] = zh;
                ztsl[dst] = zl;
            }
    }
    __syncthreads();

    // ---- GEMM2: out = conv_w @ z + conv_b, bf16-split ----
#pragma unroll
    for (int rh = 0; rh < 2; ++rh) {
        const int R = (wave * 2 + rh) * 16;
        const int row = R + lm;
        v8f acc[4];
#pragma unroll
        for (int i = 0; i < 4; ++i) acc[i] = v8f{};
        for (int k0 = 0; k0 < Cn; k0 += 32) {
            const int o1 = (k0 >> 3) + 2 * half, o2 = o1 + 1;
            v8bf h1 = *(const v8bf*)(Ch + ((size_t)o1 * 256 + row) * 8);
            v8bf h2 = *(const v8bf*)(Ch + ((size_t)o2 * 256 + row) * 8);
            v8bf l1 = *(const v8bf*)(Cl + ((size_t)o1 * 256 + row) * 8);
            v8bf l2 = *(const v8bf*)(Cl + ((size_t)o2 * 256 + row) * 8);
            v16bf ah = __builtin_shufflevector(h1, h2, 0,1,2,3,4,5,6,7,8,9,10,11,12,13,14,15);
            v16bf al = __builtin_shufflevector(l1, l2, 0,1,2,3,4,5,6,7,8,9,10,11,12,13,14,15);
            const int kb = ((k0 >> 4) + half) * 64;
#pragma unroll
            for (int ct = 0; ct < 4; ++ct) {
                const int n = ct * 16 + lm;
                v16bf bh = *(const v16bf*)(ztsh + (size_t)(kb + n) * 16);
                v16bf bl = *(const v16bf*)(ztsl + (size_t)(kb + n) * 16);
                acc[ct] = WMMAB(ah, bh, acc[ct]);
                acc[ct] = WMMAB(ah, bl, acc[ct]);
                acc[ct] = WMMAB(al, bh, acc[ct]);
            }
        }
#pragma unroll
        for (int ct = 0; ct < 4; ++ct)
#pragma unroll
            for (int i = 0; i < 8; ++i) {
                int o = R + i + 8 * half, n = ct * 16 + lm;
                out[((size_t)(b * COUT + o)) * NS + n0 + n] = acc[ct][i] + conv_b[o];
            }
    }
}

// ---------------- host launcher ----------------
extern "C" void kernel_launch(void* const* d_in, const int* in_sizes, int n_in,
                              void* d_out, int out_size, void* d_ws, size_t ws_size,
                              hipStream_t stream) {
    const float* x      = (const float*)d_in[0];
    const float* qk_w   = (const float*)d_in[1];
    const float* qk_b   = (const float*)d_in[2];
    const float* wq_w   = (const float*)d_in[3];
    const float* wq_b   = (const float*)d_in[4];
    const float* wk_w   = (const float*)d_in[5];
    const float* wk_b   = (const float*)d_in[6];
    const float* fch_w  = (const float*)d_in[7];
    const float* fch_b  = (const float*)d_in[8];
    const float* conv_w = (const float*)d_in[9];
    const float* conv_b = (const float*)d_in[10];
    float* out = (float*)d_out;
    float* ws  = (float*)d_ws;

    float*  xmean = ws + XM_OFF;
    float*  qk    = ws + QK_OFF;
    float*  qbuf  = ws + Q_OFF;
    float*  kbuf  = ws + K_OFF;
    float*  Mbuf  = ws + M_OFF;
    __bf16* Mh    = (__bf16*)(ws + MSWH_OFF);
    __bf16* Ml    = (__bf16*)(ws + MSWL_OFF);
    __bf16* Ch    = (__bf16*)(ws + CSWH_OFF);
    __bf16* Cl    = (__bf16*)(ws + CSWL_OFF);

    k_pool   <<<Bn * Cn, 256, 0, stream>>>(x, xmean);
    k_qk     <<<Bn * Cn, 256, 0, stream>>>(xmean, qk_w, qk_b, qk);
    k_heads  <<<Bn * Cn, 256, 0, stream>>>(qk, wq_w, wq_b, wk_w, wk_b, qbuf, kbuf);
    k_zero   <<<(Bn * Cn * Cn) / 256, 256, 0, stream>>>(Mbuf);
    k_attn   <<<Bn * HEADS * 4, 256, 147456, stream>>>(qbuf, kbuf, fch_w, Mbuf);
    k_attnout<<<(Bn * Cn * Cn) / 256, 256, 0, stream>>>(Mbuf, fch_b, out);
    k_splitA <<<(Bn * Cn * Cn) / 256, 256, 0, stream>>>(Mbuf, Mh, Ml);       // split M
    k_splitA <<<(Cn * Cn) / 256, 256, 0, stream>>>(conv_w, Ch, Cl);          // split conv_w
    k_biggemm<<<Bn * (NS / 64), 256, 196608, stream>>>(x, Mh, Ml, Ch, Cl,
                                                       conv_b, fch_b, out);
}
